// LSTMModel_89833535963244
// MI455X (gfx1250) — matrix operand order repaired and checked by reading, not compiled, not run on hardware
//
#include <hip/hip_runtime.h>
#include <hip/hip_bf16.h>

// ---------------- problem dims ----------------
#define Bb    512
#define Tt    40
#define Dd    300      // embed dim
#define Hh    200      // lstm dim
#define G4H   800      // 4*H
#define Vv    50257    // vocab
#define KPD   320      // Kpad for K=300 (mult of 32)
#define KPH   224      // Kpad for K=200
#define NPV   50272    // Vv padded to mult of 16
#define NPD   304      // Dd padded to mult of 16

typedef __attribute__((ext_vector_type(16))) __bf16 v16bf;
typedef __attribute__((ext_vector_type(8)))  float  v8f;
typedef __attribute__((ext_vector_type(4)))  unsigned int u32x4;

union ABf { v16bf v; unsigned int u[8]; };

__device__ __forceinline__ unsigned short f2bf(float f) {
    unsigned int u = __float_as_uint(f);
    u += 0x7FFFu + ((u >> 16) & 1u);   // round-to-nearest-even
    return (unsigned short)(u >> 16);
}

// A fragment from row-major bf16 (stride Kpad): two b128 loads.
__device__ __forceinline__ v16bf load_frag_a(const unsigned short* __restrict__ base,
                                             int Kpad, int row, int kbase, int kh) {
    const u32x4* p = (const u32x4*)(base + (long)row * Kpad + kbase + kh * 16);
    u32x4 x = p[0];               // K = kbase+kh*16 .. +7
    u32x4 y = p[1];               // K = kbase+kh*16+8 .. +15
    ABf a;
    a.u[0] = x.x; a.u[1] = x.y; a.u[2] = x.z; a.u[3] = x.w;
    a.u[4] = y.x; a.u[5] = y.y; a.u[6] = y.z; a.u[7] = y.w;
    return a.v;
}

// B fragment from column-major bf16 BT[n][k] (stride Kpad): two b128 loads.
__device__ __forceinline__ v16bf load_frag_b(const unsigned short* __restrict__ base,
                                             int Kpad, int col, int kbase, int kh) {
    const u32x4* p = (const u32x4*)(base + (long)col * Kpad + kbase + kh * 16);
    u32x4 x = p[0];               // K = kbase+kh*16 .. +7
    u32x4 y = p[1];               // K = kbase+kh*16+8 .. +15
    ABf b;
    b.u[0] = x.x; b.u[1] = x.y; b.u[2] = x.z; b.u[3] = x.w;
    b.u[4] = y.x; b.u[5] = y.y; b.u[6] = y.z; b.u[7] = y.w;
    return b.v;
}

__device__ __forceinline__ float sigm(float x) { return 1.0f / (1.0f + __expf(-x)); }

// ---------------- pack kernels (one-time, guards live here) ----------------
// out[n*Kpad + k] = bf16(in[k*Nsrc + n]) ; zero pad n>=Nsrc or k>=Ksrc
__global__ void pack_transpose_kernel(const float* __restrict__ in, unsigned short* __restrict__ out,
                                      int Ksrc, int Nsrc, int Kpad, int Npad) {
    int idx = blockIdx.x * blockDim.x + threadIdx.x;
    if (idx >= Npad * Kpad) return;
    int n = idx / Kpad, k = idx % Kpad;
    float v = (n < Nsrc && k < Ksrc) ? in[(long)k * Nsrc + n] : 0.0f;
    out[idx] = f2bf(v);
}

// out[r*Kpad + k] = bf16(in[r*Ksrc + k]) ; zero pad
__global__ void pack_rowmajor_kernel(const float* __restrict__ in, unsigned short* __restrict__ out,
                                     int Rsrc, int Ksrc, int Kpad, int Rpad) {
    int idx = blockIdx.x * blockDim.x + threadIdx.x;
    if (idx >= Rpad * Kpad) return;
    int r = idx / Kpad, k = idx % Kpad;
    float v = (r < Rsrc && k < Ksrc) ? in[(long)r * Ksrc + k] : 0.0f;
    out[idx] = f2bf(v);
}

// xbf[m*KPD + k] = bf16(emb[ids[m]*Dd + k]) ; zero pad k>=Dd
__global__ void gather_pack_kernel(const int* __restrict__ ids, const float* __restrict__ emb,
                                   unsigned short* __restrict__ xbf) {
    int idx = blockIdx.x * blockDim.x + threadIdx.x;
    if (idx >= Bb * Tt * KPD) return;
    int m = idx / KPD, k = idx % KPD;
    int token = ids[m];
    float v = (k < Dd) ? emb[(long)token * Dd + k] : 0.0f;
    xbf[idx] = f2bf(v);
}

__global__ void zero_kernel(float* __restrict__ p, int n) {
    int i = blockIdx.x * blockDim.x + threadIdx.x;
    if (i < n) p[i] = 0.0f;
}

// ---------------- xproj = xbf @ WxTbf^T + b -> f32 [T, B, 4H] ----------------
__global__ void xproj_kernel(const unsigned short* __restrict__ xbf,     // [B*T, KPD]
                             const unsigned short* __restrict__ WxTbf,   // [4H, KPD]
                             const float* __restrict__ bias,
                             float* __restrict__ xproj) {
    const int NT = G4H / 16;  // 50
    int w = (blockIdx.x * blockDim.x + threadIdx.x) >> 5;
    int lane = threadIdx.x & 31;
    int mtile = w / NT, ntile = w % NT;
    int kh = lane >> 4, nl = lane & 15;
    int arow = mtile * 16 + nl;
    int n = ntile * 16 + nl;

    v8f acc = {};
#pragma unroll
    for (int kc = 0; kc < KPD / 32; ++kc) {
        v16bf a  = load_frag_a(xbf, KPD, arow, kc * 32, kh);
        v16bf bm = load_frag_b(WxTbf, KPD, n, kc * 32, kh);
        acc = __builtin_amdgcn_wmma_f32_16x16x32_bf16(false, a, false, bm, (short)0, acc, false, false);
    }
    float bn = bias[n];
#pragma unroll
    for (int r = 0; r < 8; ++r) {
        int mm = mtile * 16 + r + 8 * kh;
        int bidx = mm / Tt, t = mm % Tt;
        xproj[((long)t * Bb + bidx) * G4H + n] = acc[r] + bn;
    }
}

// ---------------- one LSTM step ----------------
__global__ void lstm_step_kernel(const float* __restrict__ xproj_t,        // [B, 4H] f32
                                 const unsigned short* __restrict__ WhTbf, // [4H, KPH]
                                 const int* __restrict__ lengths,
                                 const unsigned short* __restrict__ hbf_in, // [B, KPH] bf16
                                 const float* __restrict__ c_in,            // [B, Hh]
                                 unsigned short* __restrict__ hbf_out,
                                 float* __restrict__ c_out,
                                 int t) {
    const int NT = (Hh + 15) / 16;  // 13
    int w = (blockIdx.x * blockDim.x + threadIdx.x) >> 5;
    if (w >= (Bb / 16) * NT) return;
    int lane = threadIdx.x & 31;
    int mtile = w / NT, htile = w % NT;
    int kh = lane >> 4, nl = lane & 15;
    int hn = htile * 16 + nl;
    int hns = (hn < Hh) ? hn : (Hh - 1);   // clamp: lanes with hn>=Hh compute garbage, never stored
    int arow = mtile * 16 + nl;

    v8f ai = {}, aj = {}, af = {}, ao = {};
#pragma unroll
    for (int kc = 0; kc < KPH / 32; ++kc) {
        int kb = kc * 32;
        v16bf a  = load_frag_a(hbf_in, KPH, arow, kb, kh);
        v16bf b0 = load_frag_b(WhTbf, KPH, 0 * Hh + hns, kb, kh);
        ai = __builtin_amdgcn_wmma_f32_16x16x32_bf16(false, a, false, b0, (short)0, ai, false, false);
        v16bf b1 = load_frag_b(WhTbf, KPH, 1 * Hh + hns, kb, kh);
        aj = __builtin_amdgcn_wmma_f32_16x16x32_bf16(false, a, false, b1, (short)0, aj, false, false);
        v16bf b2 = load_frag_b(WhTbf, KPH, 2 * Hh + hns, kb, kh);
        af = __builtin_amdgcn_wmma_f32_16x16x32_bf16(false, a, false, b2, (short)0, af, false, false);
        v16bf b3 = load_frag_b(WhTbf, KPH, 3 * Hh + hns, kb, kh);
        ao = __builtin_amdgcn_wmma_f32_16x16x32_bf16(false, a, false, b3, (short)0, ao, false, false);
    }

    if (hn < Hh) {
#pragma unroll
        for (int r = 0; r < 8; ++r) {
            int m = mtile * 16 + r + 8 * kh;
            const float* xp = xproj_t + (long)m * G4H;
            float gi = ai[r] + xp[0 * Hh + hn];
            float gj = aj[r] + xp[1 * Hh + hn];
            float gf = af[r] + xp[2 * Hh + hn];
            float go = ao[r] + xp[3 * Hh + hn];
            float cold = c_in[(long)m * Hh + hn];
            float cn = sigm(gf + 1.0f) * cold + sigm(gi) * tanhf(gj);
            float hv = sigm(go) * tanhf(cn);
            bool msk = t < lengths[m];
            c_out[(long)m * Hh + hn] = msk ? cn : cold;
            hbf_out[(long)m * KPH + hn] = msk ? f2bf(hv) : hbf_in[(long)m * KPH + hn];
        }
    }
}

// ---------------- h1bf = bf16(relu(c @ U + b1)) : [512, KPD] ----------------
__global__ void h1_kernel(const unsigned short* __restrict__ cbf,   // [B, KPH]
                          const unsigned short* __restrict__ UTbf,  // [NPD, KPH] (rows >= Dd are zero)
                          const float* __restrict__ b1,
                          unsigned short* __restrict__ h1bf) {      // [B, KPD] pre-zeroed
    const int NT = NPD / 16;  // 19
    int w = (blockIdx.x * blockDim.x + threadIdx.x) >> 5;
    if (w >= (Bb / 16) * NT) return;
    int lane = threadIdx.x & 31;
    int mtile = w / NT, ntile = w % NT;
    int kh = lane >> 4, nl = lane & 15;
    int n = ntile * 16 + nl;
    int arow = mtile * 16 + nl;

    v8f acc = {};
#pragma unroll
    for (int kc = 0; kc < KPH / 32; ++kc) {
        v16bf a  = load_frag_a(cbf, KPH, arow, kc * 32, kh);
        v16bf bm = load_frag_b(UTbf, KPH, n, kc * 32, kh);
        acc = __builtin_amdgcn_wmma_f32_16x16x32_bf16(false, a, false, bm, (short)0, acc, false, false);
    }
    if (n < Dd) {
        float bn = b1[n];
#pragma unroll
        for (int r = 0; r < 8; ++r) {
            int m = mtile * 16 + r + 8 * kh;
            float v = acc[r] + bn;
            h1bf[(long)m * KPD + n] = f2bf(v > 0.0f ? v : 0.0f);
        }
    }
}

// ---------------- logits = h1 @ emb.T + b2 : f32 [512, Vv] ----------------
__global__ void logits_kernel(const unsigned short* __restrict__ h1bf,  // [B, KPD]
                              const unsigned short* __restrict__ embbf, // [NPV, KPD] (rows >= Vv zero)
                              const float* __restrict__ b2,
                              float* __restrict__ out) {
    const int NT = NPV / 16;  // 3142
    int w = (blockIdx.x * blockDim.x + threadIdx.x) >> 5;
    if (w >= (Bb / 16) * NT) return;
    int lane = threadIdx.x & 31;
    int mtile = w / NT, ntile = w % NT;
    int kh = lane >> 4, nl = lane & 15;
    int n = ntile * 16 + nl;
    int arow = mtile * 16 + nl;

    v8f acc = {};
#pragma unroll
    for (int kc = 0; kc < KPD / 32; ++kc) {
        v16bf a  = load_frag_a(h1bf, KPD, arow, kc * 32, kh);
        v16bf bm = load_frag_b(embbf, KPD, n, kc * 32, kh);
        acc = __builtin_amdgcn_wmma_f32_16x16x32_bf16(false, a, false, bm, (short)0, acc, false, false);
    }
    if (n < Vv) {
        float bn = b2[n];
#pragma unroll
        for (int r = 0; r < 8; ++r) {
            int m = mtile * 16 + r + 8 * kh;
            out[(long)m * Vv + n] = acc[r] + bn;
        }
    }
}

// ---------------- launch ----------------
extern "C" void kernel_launch(void* const* d_in, const int* in_sizes, int n_in,
                              void* d_out, int out_size, void* d_ws, size_t ws_size,
                              hipStream_t stream) {
    (void)in_sizes; (void)n_in; (void)out_size; (void)ws_size;
    const int*   ids     = (const int*)d_in[0];
    const int*   lengths = (const int*)d_in[1];
    const float* emb     = (const float*)d_in[2];
    const float* Wx      = (const float*)d_in[3];
    const float* Wh      = (const float*)d_in[4];
    const float* b       = (const float*)d_in[5];
    const float* U       = (const float*)d_in[6];
    const float* b1      = (const float*)d_in[7];
    const float* b2      = (const float*)d_in[8];
    float*       out     = (float*)d_out;

    // workspace carve (all chunk sizes multiples of 16 bytes)
    char* p = (char*)d_ws;
    unsigned short* xbf   = (unsigned short*)p; p += (long)Bb * Tt * KPD * 2;  // 13.1 MB
    unsigned short* WxTbf = (unsigned short*)p; p += (long)G4H * KPD * 2;      // 512 KB
    unsigned short* WhTbf = (unsigned short*)p; p += (long)G4H * KPH * 2;      // 358 KB
    unsigned short* UTbf  = (unsigned short*)p; p += (long)NPD * KPH * 2;      // 136 KB
    unsigned short* embbf = (unsigned short*)p; p += (long)NPV * KPD * 2;      // 32.2 MB
    unsigned short* hbf0  = (unsigned short*)p; p += (long)Bb * KPH * 2;
    unsigned short* hbf1  = (unsigned short*)p; p += (long)Bb * KPH * 2;
    unsigned short* cbf   = (unsigned short*)p; p += (long)Bb * KPH * 2;
    unsigned short* h1bf  = (unsigned short*)p; p += (long)Bb * KPD * 2;
    float* c0    = (float*)p; p += (long)Bb * Hh * 4;
    float* c1    = (float*)p; p += (long)Bb * Hh * 4;
    float* xproj = (float*)p; p += (long)Tt * Bb * G4H * 4;                    // 65.5 MB

    const int TPB = 256;
    auto blocks = [](long n) { return (int)((n + 255) / 256); };

    // 1) one-time packs (transpose + f32->bf16 + padding)
    pack_transpose_kernel<<<blocks((long)G4H * KPD), TPB, 0, stream>>>(Wx, WxTbf, Dd, G4H, KPD, G4H);
    pack_transpose_kernel<<<blocks((long)G4H * KPH), TPB, 0, stream>>>(Wh, WhTbf, Hh, G4H, KPH, G4H);
    pack_transpose_kernel<<<blocks((long)NPD * KPH), TPB, 0, stream>>>(U, UTbf, Hh, Dd, KPH, NPD);
    pack_rowmajor_kernel<<<blocks((long)NPV * KPD), TPB, 0, stream>>>(emb, embbf, Vv, Dd, KPD, NPV);
    gather_pack_kernel<<<blocks((long)Bb * Tt * KPD), TPB, 0, stream>>>(ids, emb, xbf);

    // 2) zero-init: h double buffers (bf16, incl. padding), c0, h1bf padding
    zero_kernel<<<blocks((long)Bb * KPH), TPB, 0, stream>>>((float*)hbf0, Bb * KPH);     // covers hbf0+hbf1 (2B each)
    zero_kernel<<<blocks((long)Bb * Hh), TPB, 0, stream>>>(c0, Bb * Hh);
    zero_kernel<<<blocks((long)Bb * KPD / 2), TPB, 0, stream>>>((float*)h1bf, Bb * KPD / 2);

    // 3) input projection GEMM: 1280 x 50 wave tiles
    xproj_kernel<<<(Bb * Tt / 16) * (G4H / 16) / 8, TPB, 0, stream>>>(xbf, WxTbf, b, xproj);

    // 4) 40 sequential LSTM steps (double-buffered h bf16 / c f32)
    for (int t = 0; t < Tt; ++t) {
        const unsigned short* hin = (t & 1) ? hbf1 : hbf0;
        unsigned short*       hout = (t & 1) ? hbf0 : hbf1;
        const float* cin = (t & 1) ? c1 : c0;
        float*       cout = (t & 1) ? c0 : c1;
        lstm_step_kernel<<<52, TPB, 0, stream>>>(xproj + (long)t * Bb * G4H, WhTbf,
                                                 lengths, hin, cin, hout, cout, t);
    }
    // after t=39 (odd) final c is in c0
    pack_rowmajor_kernel<<<blocks((long)Bb * KPH), TPB, 0, stream>>>(c0, cbf, Bb, Hh, KPH, Bb);

    // 5) h1 = relu(c @ U + b1) -> bf16
    h1_kernel<<<(Bb / 16) * (NPD / 16) / 8, TPB, 0, stream>>>(cbf, UTbf, b1, h1bf);

    // 6) logits = h1 @ emb.T + b2
    logits_kernel<<<(Bb / 16) * (NPV / 16) / 8, TPB, 0, stream>>>(h1bf, embbf, b2, out);
}
